// RNNModel_37434934952280
// MI455X (gfx1250) — matrix operand order repaired and checked by reading, not compiled, not run on hardware
//
#include <hip/hip_runtime.h>
#include <math.h>

#define TT    128
#define BB    16
#define NHID  1024
#define NTOK  32000
#define NSTEP 8

typedef __attribute__((ext_vector_type(16))) __bf16 v16bf;
typedef __attribute__((ext_vector_type(8)))  float  v8f;

__constant__ int d_pred[8] = {0, 0, 1, 2, 3, 4, 5, 6};
__constant__ int d_act[8]  = {0, 1, 2, 3, 0, 1, 2, 3};

__device__ __forceinline__ float act_f(int a, float x) {
    switch (a) {
        case 0:  return tanhf(x);
        case 1:  return fmaxf(x, 0.f);
        case 2:  return 1.f / (1.f + __expf(-x));
        default: return x;
    }
}

// ---------------------------------------------------------------------------
// Pack W0 [2048,2048] f32 (row=k, col=n) into bf16 WMMA B-fragments.
// Fragment layout (contiguous k, same order as the A packing): lane l -> col n=(l&15),
// k-half = l>>4 (lanes 0-15: K 0-15, lanes 16-31: K 16-31), elem e -> k offset.
// packed[((ntile*Kf + kf)*32 + l)*16 + e] = W[k, n]
// ---------------------------------------------------------------------------
__global__ void pack_w0_kernel(const float* __restrict__ W0, __bf16* __restrict__ pW0) {
    size_t i = (size_t)blockIdx.x * blockDim.x + threadIdx.x;   // 2048*2048
    if (i >= (size_t)2048 * 2048) return;
    int e  = (int)(i & 15);
    int l  = (int)((i >> 4) & 31);
    size_t fr = i >> 9;
    int kf = (int)(fr & 63);          // Kf = 64
    int nt = (int)(fr >> 6);
    int k = kf * 32 + ((l >> 4) << 4) + e;
    int n = nt * 16 + (l & 15);
    pW0[i] = (__bf16)W0[(size_t)k * 2048 + n];
}

__global__ void pack_ws_kernel(const float* __restrict__ Ws, __bf16* __restrict__ pWs) {
    size_t i = (size_t)blockIdx.x * blockDim.x + threadIdx.x;   // 8*1024*2048
    const size_t per = (size_t)NHID * 2 * NHID;
    if (i >= (size_t)NSTEP * per) return;
    int s = (int)(i / per);
    size_t j = i % per;
    int e  = (int)(j & 15);
    int l  = (int)((j >> 4) & 31);
    size_t fr = j >> 9;
    int kf = (int)(fr & 31);          // Kf = 32
    int nt = (int)(fr >> 5);
    int k = kf * 32 + ((l >> 4) << 4) + e;
    int n = nt * 16 + (l & 15);
    pWs[i] = (__bf16)Ws[(size_t)s * per + (size_t)k * 2048 + n];
}

// ---------------------------------------------------------------------------
// Pack decoder B = emb^T into bf16 B-fragments.
// B[k, n] = emb[n, k]; per lane the 16 source floats are contiguous in k.
// packedE[((ntile*32 + kf)*32 + l)*16 + e] = emb[n*NHID + k]
// ---------------------------------------------------------------------------
__global__ void pack_emb_kernel(const float* __restrict__ emb, __bf16* __restrict__ pE) {
    size_t i = (size_t)blockIdx.x * blockDim.x + threadIdx.x;   // NTOK*NHID
    if (i >= (size_t)NTOK * NHID) return;
    int e  = (int)(i & 15);
    int l  = (int)((i >> 4) & 31);
    size_t fr = i >> 9;
    int kf = (int)(fr & 31);          // Kf = 32
    int nt = (int)(fr >> 5);          // 0..1999
    int k = kf * 32 + ((l >> 4) << 4) + e;
    int n = nt * 16 + (l & 15);
    pE[i] = (__bf16)emb[(size_t)n * NHID + k];
}

// ---------------------------------------------------------------------------
// Embedding gather: x_emb[(t*B+b), k] = emb[tokens[t*B+b], k]
// ---------------------------------------------------------------------------
__global__ void embed_kernel(const int* __restrict__ tokens, const float* __restrict__ emb,
                             float* __restrict__ x_emb) {
    size_t i = (size_t)blockIdx.x * blockDim.x + threadIdx.x;   // T*B*NHID
    if (i >= (size_t)TT * BB * NHID) return;
    size_t row = i >> 10;
    int k = (int)(i & (NHID - 1));
    x_emb[i] = emb[(size_t)tokens[row] * NHID + k];
}

// ---------------------------------------------------------------------------
// One recurrence timestep. Single workgroup, 1024 threads = 32 wave32 waves.
// Stage -1: ch0 = [x_t, h_prev] @ W0 (K=2048), combine with tanh -> states[0]
// Stage s (0..7): states[pred] @ Ws[s] (K=1024) -> states[s+1]; mean -> hiddens[t]
// A-matrix fragments (contiguous k: lanes 0-15 K 0-15,
// lanes 16-31 K 16-31) built in LDS, shared by all waves.
// Wave w owns c-tile n_tile=jc and h-tile n_tile=jc+64 so the gated combine
// happens on matching register tiles in the WMMA D layout.
// ---------------------------------------------------------------------------
__global__ __launch_bounds__(1024) void cell_kernel(
    const float* __restrict__ x_emb, const float* __restrict__ h0,
    float* __restrict__ hiddens,
    const __bf16* __restrict__ pW0, const __bf16* __restrict__ pWs,
    float* __restrict__ states, int t)
{
    extern __shared__ char smem[];
    __bf16* ldsA   = (__bf16*)smem;             // up to 64 frags * 512 bf16 = 64 KB
    float*  sumbuf = (float*)(smem + 65536);    // 16*1024 f32 = 64 KB

    const int tid  = threadIdx.x;
    const int lane = tid & 31;
    const int wave = tid >> 5;

    for (int i = tid; i < BB * NHID; i += 1024) sumbuf[i] = 0.f;

    const float* hprev = (t == 0) ? h0 : hiddens + (size_t)(t - 1) * BB * NHID;
    const float* xt    = x_emb + (size_t)t * BB * NHID;

    for (int stage = -1; stage < NSTEP; ++stage) {
        const float*  sprev;
        const __bf16* pB;
        int Kf, actI;
        if (stage < 0) { sprev = hprev; pB = pW0; Kf = 64; actI = 0; }
        else {
            sprev = states + (size_t)d_pred[stage] * BB * NHID;
            pB    = pWs + (size_t)stage * NHID * 2 * NHID;
            Kf    = 32;
            actI  = d_act[stage];
        }

        // Pack A (16 x Kf*32) into LDS fragment layout (bf16).
        for (int i = tid; i < Kf * 512; i += 1024) {
            int e  = i & 15;
            int l  = (i >> 4) & 31;
            int kf = i >> 9;
            int m  = l & 15;
            int k  = kf * 32 + ((l >> 4) << 4) + e;
            float v;
            if (stage < 0) v = (k < NHID) ? xt[m * NHID + k] : hprev[m * NHID + (k - NHID)];
            else           v = sprev[m * NHID + k];
            ldsA[i] = (__bf16)v;
        }
        __syncthreads();

        float* sout = states + (size_t)(stage + 1) * BB * NHID;
        const v16bf* aF = (const v16bf*)ldsA;
        const v16bf* bF = (const v16bf*)pB;

        for (int p = 0; p < 2; ++p) {
            const int jc = wave + 32 * p;   // c-half n-tile (0..63)
            const int jh = jc + 64;         // matching h-half n-tile
            v8f accC = {}, accH = {};
            for (int kf = 0; kf < Kf; ++kf) {
                v16bf a  = aF[kf * 32 + lane];
                v16bf bc = bF[(size_t)(jc * Kf + kf) * 32 + lane];
                v16bf bh = bF[(size_t)(jh * Kf + kf) * 32 + lane];
                accC = __builtin_amdgcn_wmma_f32_16x16x32_bf16(false, a, false, bc,
                                                               (short)0, accC, false, false);
                accH = __builtin_amdgcn_wmma_f32_16x16x32_bf16(false, a, false, bh,
                                                               (short)0, accH, false, false);
            }
            // Gated combine on WMMA D layout: VGPR r -> row, lane -> col.
            const int n  = jc * 16 + (lane & 15);
            const int mb = (lane >> 4) << 3;
            #pragma unroll
            for (int r = 0; r < 8; ++r) {
                int   m  = mb + r;
                float sp = sprev[m * NHID + n];
                float sg = 1.f / (1.f + __expf(-accC[r]));
                float s  = sp + sg * (act_f(actI, accH[r]) - sp);
                sout[m * NHID + n] = s;
                if (stage >= 0) sumbuf[m * NHID + n] += s;
            }
        }
        __threadfence();
        __syncthreads();
    }

    float* hout = hiddens + (size_t)t * BB * NHID;
    for (int i = tid; i < BB * NHID; i += 1024) hout[i] = sumbuf[i] * 0.125f;
}

// ---------------------------------------------------------------------------
// Decoder: logits[r, n] = sum_k hiddens[r,k] * emb[n,k] + dec_b[n]
// Block = 256 threads (8 waves), handles one 16-row block and 8 n-tiles.
// A (16x1024 hidden block) packed to LDS bf16 fragments; B streamed from the
// pre-packed bf16 emb fragments (L2-resident): 2x global_load_b128 + 1 wmma
// per fragment per lane.
// ---------------------------------------------------------------------------
__global__ __launch_bounds__(256) void decoder_kernel(
    const float* __restrict__ hiddens, const __bf16* __restrict__ pE,
    const float* __restrict__ dec_b, float* __restrict__ out)
{
    const int rowblk = blockIdx.y;            // 0..127
    const int lane   = threadIdx.x & 31;
    const int wave   = threadIdx.x >> 5;

    __shared__ __align__(32) __bf16 ldsA[32 * 512];   // 32 KB
    const float* Arows = hiddens + (size_t)rowblk * 16 * NHID;
    for (int i = threadIdx.x; i < 32 * 512; i += 256) {
        int e  = i & 15;
        int l  = (i >> 4) & 31;
        int kf = i >> 9;
        int m  = l & 15;
        int k  = kf * 32 + ((l >> 4) << 4) + e;
        ldsA[i] = (__bf16)Arows[m * NHID + k];
    }
    __syncthreads();

    const int ntile = blockIdx.x * 8 + wave;  // 0..1999
    const int ncol  = ntile * 16 + (lane & 15);

    v8f acc = {};
    const v16bf* aF = (const v16bf*)ldsA;
    const v16bf* bF = (const v16bf*)pE + (size_t)ntile * 32 * 32;  // 32 frags * 32 lanes
    for (int kf = 0; kf < 32; ++kf) {
        v16bf a = aF[kf * 32 + lane];
        __builtin_prefetch((const void*)(bF + (kf + 4) * 32 + lane), 0, 1); // global_prefetch_b8
        v16bf b = bF[kf * 32 + lane];
        acc = __builtin_amdgcn_wmma_f32_16x16x32_bf16(false, a, false, b,
                                                      (short)0, acc, false, false);
    }

    const float bias = dec_b[ncol];
    const int   mb   = (lane >> 4) << 3;
    #pragma unroll
    for (int r = 0; r < 8; ++r) {
        int m = mb + r;
        out[(size_t)(rowblk * 16 + m) * NTOK + ncol] = acc[r] + bias;
    }
}

// ---------------------------------------------------------------------------
// log_softmax: pass 1 per-row max + log-sum-exp; pass 2 apply.
// ---------------------------------------------------------------------------
__global__ void softmax_reduce_kernel(const float* __restrict__ logits,
                                      float* __restrict__ stats)
{
    const int r = blockIdx.x;                 // 0..2047
    const float* row = logits + (size_t)r * NTOK;
    __shared__ float red[256];

    float mx = -INFINITY;
    for (int i = threadIdx.x; i < NTOK; i += 256) mx = fmaxf(mx, row[i]);
    red[threadIdx.x] = mx;
    __syncthreads();
    for (int s = 128; s > 0; s >>= 1) {
        if (threadIdx.x < s) red[threadIdx.x] = fmaxf(red[threadIdx.x], red[threadIdx.x + s]);
        __syncthreads();
    }
    mx = red[0];
    __syncthreads();

    float sum = 0.f;
    for (int i = threadIdx.x; i < NTOK; i += 256) sum += __expf(row[i] - mx);
    red[threadIdx.x] = sum;
    __syncthreads();
    for (int s = 128; s > 0; s >>= 1) {
        if (threadIdx.x < s) red[threadIdx.x] += red[threadIdx.x + s];
        __syncthreads();
    }
    if (threadIdx.x == 0) { stats[r * 2] = mx; stats[r * 2 + 1] = logf(red[0]); }
}

__global__ void softmax_apply_kernel(float* __restrict__ out,
                                     const float* __restrict__ stats)
{
    const int r = blockIdx.y;                                     // 0..2047
    const int c = blockIdx.x * blockDim.x + threadIdx.x;          // 0..31999
    if (c >= NTOK) return;
    size_t i = (size_t)r * NTOK + c;
    out[i] = out[i] - stats[r * 2] - stats[r * 2 + 1];
}

__global__ void copy_hlast_kernel(const float* __restrict__ hiddens,
                                  float* __restrict__ out)
{
    int i = blockIdx.x * blockDim.x + threadIdx.x;                // B*NHID
    if (i >= BB * NHID) return;
    out[i] = hiddens[(size_t)(TT - 1) * BB * NHID + i];
}

// ---------------------------------------------------------------------------
extern "C" void kernel_launch(void* const* d_in, const int* in_sizes, int n_in,
                              void* d_out, int out_size, void* d_ws, size_t ws_size,
                              hipStream_t stream) {
    const int*   tokens  = (const int*)d_in[0];
    const float* hidden0 = (const float*)d_in[1];
    const float* emb     = (const float*)d_in[2];
    const float* W0      = (const float*)d_in[3];
    const float* Ws      = (const float*)d_in[4];
    const float* dec_b   = (const float*)d_in[5];
    float* out = (float*)d_out;
    char*  ws  = (char*)d_ws;

    // Workspace layout (~119 MB total)
    size_t off = 0;
    __bf16* pW0 = (__bf16*)(ws + off); off += (size_t)2048 * 2048 * 2;          // 8 MB
    __bf16* pWs = (__bf16*)(ws + off); off += (size_t)NSTEP * NHID * 2 * NHID * 2; // 32 MB
    __bf16* pE  = (__bf16*)(ws + off); off += (size_t)NTOK * NHID * 2;          // 62.5 MB
    float* xemb  = (float*)(ws + off); off += (size_t)TT * BB * NHID * 4;       // 8 MB
    float* hid   = (float*)(ws + off); off += (size_t)TT * BB * NHID * 4;       // 8 MB
    float* state = (float*)(ws + off); off += (size_t)9 * BB * NHID * 4;        // 576 KB
    float* stats = (float*)(ws + off);                                          // 16 KB

    // 1) Pack weights + tied decoder matrix to bf16 WMMA fragment layout
    pack_w0_kernel<<<(2048 * 2048) / 256, 256, 0, stream>>>(W0, pW0);
    pack_ws_kernel<<<(NSTEP * NHID * 2 * NHID) / 256, 256, 0, stream>>>(Ws, pWs);
    pack_emb_kernel<<<(int)(((size_t)NTOK * NHID) / 256), 256, 0, stream>>>(emb, pE);

    // 2) Embedding gather
    embed_kernel<<<(TT * BB * NHID) / 256, 256, 0, stream>>>(tokens, emb, xemb);

    // 3) Sequential recurrence: one launch per timestep is the scan's sync
    for (int t = 0; t < TT; ++t) {
        cell_kernel<<<1, 1024, 131072, stream>>>(xemb, hidden0, hid, pW0, pWs, state, t);
    }

    // 4) Tied decoder GEMM: 2048 x 32000 logits into d_out
    decoder_kernel<<<dim3(NTOK / (16 * 8), (TT * BB) / 16), 256, 0, stream>>>(
        hid, pE, dec_b, out);

    // 5) h_last tail output
    copy_hlast_kernel<<<(BB * NHID) / 256, 256, 0, stream>>>(
        hid, out + (size_t)TT * BB * NTOK);

    // 6) log_softmax in place over the logits region
    softmax_reduce_kernel<<<TT * BB, 256, 0, stream>>>(out, stats);
    softmax_apply_kernel<<<dim3(NTOK / 256, TT * BB), 256, 0, stream>>>(out, stats);
}
